// ProceduralMemory_20976620274276
// MI455X (gfx1250) — matrix operand order repaired and checked by reading, not compiled, not run on hardware
//
#include <hip/hip_runtime.h>

typedef __attribute__((ext_vector_type(16))) __bf16 v16bf;
typedef __attribute__((ext_vector_type(8)))  __bf16 v8bf;
typedef __attribute__((ext_vector_type(8)))  float  v8f;

#define D_DIM   1024
#define BS      32
#define P_TOK   128
#define R_SLOT  16
#define M_TOTAL (BS * P_TOK)   // 4096
#define RHO_C   0.97f
#define EPS_C   1e-8f
#define LDS_PITCH 40           // bf16 elements per row (80 B): 16B-aligned, bank-spread

union V16 { v16bf v; v8bf h2[2]; };

// Split fp32 -> (bf16 hi, bf16 lo): hi+lo keeps ~16 mantissa bits.
__device__ __forceinline__ void split16(const float* v, v16bf& h, v16bf& l) {
#pragma unroll
  for (int j = 0; j < 16; ++j) {
    __bf16 hb = (__bf16)v[j];
    h[j] = hb;
    l[j] = (__bf16)(v[j] - (float)hb);
  }
}

// A fragment (16-bit A 16x32, ISA 7.12.2): lane-half h holds row m,
// elements 0..7 = K 8h..8h+7, elements 8..15 = K 16+8h..16+8h+7.
__device__ __forceinline__ v16bf ldsA(const __bf16 (*s)[LDS_PITCH], int m, int half) {
  V16 u;
  u.h2[0] = *(const v8bf*)&s[m][8 * half];
  u.h2[1] = *(const v8bf*)&s[m][16 + 8 * half];
  return u.v;
}

// B fragment (16-bit B 32x16): lane holds column n, elements 0..7 = K 8h..8h+7, elements 8..15 = K 16+8h..16+8h+7.
__device__ __forceinline__ v16bf ldsB(const __bf16 (*s)[LDS_PITCH], int n, int half) {
  V16 u;
  u.h2[0] = *(const v8bf*)&s[n][8 * half];
  u.h2[1] = *(const v8bf*)&s[n][16 + 8 * half];
  return u.v;
}

// ---------------------------------------------------------------------------
// Kernel 1: C = A[4096x1024] * W[1024x1024] + bias, bf16x3 WMMA (f32 accum).
// WG tile 128x128, 8 waves (4Mx2N), wave tile 32x64 = 2x4 accumulators.
// Both operands staged in LDS as pre-split bf16 hi/lo once per workgroup.
// blockIdx.z selects (x_all,W_k,b_k)->Kraw vs (h_all,W_v,b_v)->Vraw.
// ---------------------------------------------------------------------------
__global__ __launch_bounds__(256)
void gemm_bf16x3_kernel(const float* __restrict__ A0, const float* __restrict__ W0,
                        const float* __restrict__ bias0, float* __restrict__ out0,
                        const float* __restrict__ A1, const float* __restrict__ W1,
                        const float* __restrict__ bias1, float* __restrict__ out1)
{
  const float* A   = (blockIdx.z == 0) ? A0    : A1;
  const float* W   = (blockIdx.z == 0) ? W0    : W1;
  const float* bia = (blockIdx.z == 0) ? bias0 : bias1;
  float*       out = (blockIdx.z == 0) ? out0  : out1;

  __shared__ __align__(16) __bf16 sAh[128][LDS_PITCH];  // [m][k]  A hi
  __shared__ __align__(16) __bf16 sAl[128][LDS_PITCH];  // [m][k]  A lo
  __shared__ __align__(16) __bf16 sBh[128][LDS_PITCH];  // [n][k]  W hi (transposed)
  __shared__ __align__(16) __bf16 sBl[128][LDS_PITCH];  // [n][k]  W lo

  const int tid  = threadIdx.x;
  const int lane = tid & 31;
  const int wave = tid >> 5;
  const int wm   = wave >> 1;      // 0..3
  const int wn   = wave & 1;       // 0..1
  const int half = lane >> 4;      // 0..1
  const int l16  = lane & 15;

  const int m0 = blockIdx.y * 128;
  const int n0 = blockIdx.x * 128;

  // Staging assignment: each thread owns 16 contiguous k's of one row/column.
  const int st_row = tid >> 1;          // 0..127 (m for A, n for B)
  const int st_kb  = (tid & 1) * 16;    // 0 or 16

  v8f acc[2][4] = {};

  for (int kk = 0; kk < D_DIM; kk += 32) {
    __syncthreads();   // previous iteration's LDS reads complete

    // ---- Stage A tile [128 x 32]: contiguous 64B reads per thread ----
    {
      const float* ap = A + (size_t)(m0 + st_row) * D_DIM + kk + st_kb;
      float av[16];
#pragma unroll
      for (int j = 0; j < 16; ++j) av[j] = ap[j];
      v16bf hi, lo; split16(av, hi, lo);
      V16 uh, ul; uh.v = hi; ul.v = lo;
      *(v8bf*)&sAh[st_row][st_kb]     = uh.h2[0];
      *(v8bf*)&sAh[st_row][st_kb + 8] = uh.h2[1];
      *(v8bf*)&sAl[st_row][st_kb]     = ul.h2[0];
      *(v8bf*)&sAl[st_row][st_kb + 8] = ul.h2[1];
    }
    // ---- Stage W tile [32 x 128] transposed -> [n][k] ----
    {
      const float* wp = W + (size_t)(kk + st_kb) * D_DIM + n0 + st_row;
      float bv[16];
#pragma unroll
      for (int j = 0; j < 16; ++j) bv[j] = wp[(size_t)j * D_DIM];  // column walk
      v16bf hi, lo; split16(bv, hi, lo);
      V16 uh, ul; uh.v = hi; ul.v = lo;
      *(v8bf*)&sBh[st_row][st_kb]     = uh.h2[0];
      *(v8bf*)&sBh[st_row][st_kb + 8] = uh.h2[1];
      *(v8bf*)&sBl[st_row][st_kb]     = ul.h2[0];
      *(v8bf*)&sBl[st_row][st_kb + 8] = ul.h2[1];
    }
    __syncthreads();

    // ---- Fragments from LDS + 3-WMMA split-precision products ----
    v16bf ah[2], al[2];
#pragma unroll
    for (int mi = 0; mi < 2; ++mi) {
      const int m = wm * 32 + mi * 16 + l16;
      ah[mi] = ldsA(sAh, m, half);
      al[mi] = ldsA(sAl, m, half);
    }
#pragma unroll
    for (int ni = 0; ni < 4; ++ni) {
      const int n = wn * 64 + ni * 16 + l16;
      v16bf bh = ldsB(sBh, n, half);
      v16bf bl = ldsB(sBl, n, half);
#pragma unroll
      for (int mi = 0; mi < 2; ++mi) {
        acc[mi][ni] = __builtin_amdgcn_wmma_f32_16x16x32_bf16(
            false, ah[mi], false, bh, (short)0, acc[mi][ni], false, false);
        acc[mi][ni] = __builtin_amdgcn_wmma_f32_16x16x32_bf16(
            false, ah[mi], false, bl, (short)0, acc[mi][ni], false, false);
        acc[mi][ni] = __builtin_amdgcn_wmma_f32_16x16x32_bf16(
            false, al[mi], false, bh, (short)0, acc[mi][ni], false, false);
      }
    }
  }

  // Epilogue: +bias, store. C layout: lane -> N = l16, M = i + 8*half.
#pragma unroll
  for (int ni = 0; ni < 4; ++ni) {
    const int colc = n0 + wn * 64 + ni * 16 + l16;
    const float bv = bia[colc];
#pragma unroll
    for (int mi = 0; mi < 2; ++mi) {
#pragma unroll
      for (int i = 0; i < 8; ++i) {
        const int rowc = m0 + wm * 32 + mi * 16 + half * 8 + i;
        out[(size_t)rowc * D_DIM + colc] = acc[mi][ni][i] + bv;
      }
    }
  }
}

// ---------------------------------------------------------------------------
// Kernel 2: eligibility. The P-scan collapses analytically:
// c[p] = g[p] * prod_{q>p} a[q];
// elig[b,r,d] = Atot*e0[b,r,d] + sum_p c[p] * (normalized k / raw v)[p,d].
// ---------------------------------------------------------------------------
__global__ __launch_bounds__(256)
void elig_kernel(const float* __restrict__ Kraw, const float* __restrict__ Vraw,
                 const float* __restrict__ surprise, const int* __restrict__ reset,
                 const float* __restrict__ eK0, const float* __restrict__ eV0,
                 float* __restrict__ outK, float* __restrict__ outV)
{
  const int b = blockIdx.x;
  __shared__ float s_coef[P_TOK];   // g[p] * suffix-prod(a)
  __shared__ float s_ck[P_TOK];     // s_coef[p] / max(||k_raw[p]||, eps)
  __shared__ float s_at;

  if (threadIdx.x == 0) {
    float suffix = 1.0f;
    for (int p = P_TOK - 1; p >= 0; --p) {
      float g = fminf(fmaxf(surprise[b * P_TOK + p] * 0.2f, 0.0f), 1.0f);
      s_coef[p] = g * suffix;
      float a = RHO_C * (reset[b * P_TOK + p] ? 0.0f : 1.0f);
      suffix *= a;
    }
    s_at = suffix;
  }
  __syncthreads();

  const int lane = threadIdx.x & 31;
  const int wave = threadIdx.x >> 5;

  for (int p = wave; p < P_TOK; p += 8) {
    const float* kr = Kraw + ((size_t)b * P_TOK + p) * D_DIM;
    float ss = 0.0f;
    for (int d = lane; d < D_DIM; d += 32) { float v = kr[d]; ss += v * v; }
#pragma unroll
    for (int off = 16; off > 0; off >>= 1) ss += __shfl_xor(ss, off);
    float inv = 1.0f / fmaxf(sqrtf(ss), EPS_C);
    if (lane == 0) s_ck[p] = s_coef[p] * inv;
  }
  __syncthreads();

  const float at = s_at;
  for (int d = threadIdx.x; d < D_DIM; d += 256) {
    float sk = 0.0f, sv = 0.0f;
    for (int p = 0; p < P_TOK; ++p) {
      sk += s_ck[p]   * Kraw[((size_t)b * P_TOK + p) * D_DIM + d];
      sv += s_coef[p] * Vraw[((size_t)b * P_TOK + p) * D_DIM + d];
    }
#pragma unroll
    for (int r = 0; r < R_SLOT; ++r) {
      const size_t idx = ((size_t)b * R_SLOT + r) * D_DIM + d;
      outK[idx] = at * eK0[idx] + sk;
      outV[idx] = at * eV0[idx] + sv;
    }
  }
}

// ---------------------------------------------------------------------------
// Kernel 3: slot-memory readout. One wave per token.
// ---------------------------------------------------------------------------
__global__ __launch_bounds__(256)
void readout_kernel(const float* __restrict__ x_all, const float* __restrict__ pm_K,
                    const float* __restrict__ pm_V, const float* __restrict__ pm_a,
                    float* __restrict__ y_out)
{
  const int b    = blockIdx.x;
  const int lane = threadIdx.x & 31;
  const int wave = threadIdx.x >> 5;
  const int p    = blockIdx.y * 8 + wave;

  const float* x = x_all + ((size_t)b * P_TOK + p) * D_DIM;
  float xr[32];
  float ss = 0.0f;
#pragma unroll
  for (int i = 0; i < 32; ++i) { xr[i] = x[lane + 32 * i]; ss += xr[i] * xr[i]; }
#pragma unroll
  for (int off = 16; off > 0; off >>= 1) ss += __shfl_xor(ss, off);
  const float inv = 1.0f / fmaxf(sqrtf(ss), EPS_C);

  float w[R_SLOT];
#pragma unroll
  for (int r = 0; r < R_SLOT; ++r) {
    const float* kr = pm_K + ((size_t)b * R_SLOT + r) * D_DIM;
    float s = 0.0f;
#pragma unroll 8
    for (int i = 0; i < 32; ++i) s += kr[lane + 32 * i] * xr[i];
#pragma unroll
    for (int off = 16; off > 0; off >>= 1) s += __shfl_xor(s, off);
    w[r] = pm_a[b * R_SLOT + r] * s * inv;
  }

  float* y = y_out + ((size_t)b * P_TOK + p) * D_DIM;
#pragma unroll 4
  for (int i = 0; i < 32; ++i) {
    const int d = lane + 32 * i;
    float acc = 0.0f;
#pragma unroll
    for (int r = 0; r < R_SLOT; ++r)
      acc += w[r] * pm_V[((size_t)b * R_SLOT + r) * D_DIM + d];
    y[d] = acc;
  }
}

// ---------------------------------------------------------------------------
extern "C" void kernel_launch(void* const* d_in, const int* in_sizes, int n_in,
                              void* d_out, int out_size, void* d_ws, size_t ws_size,
                              hipStream_t stream) {
  const float* x_all    = (const float*)d_in[0];
  const float* h_all    = (const float*)d_in[1];
  const float* surprise = (const float*)d_in[2];
  const int*   reset    = (const int*)d_in[3];
  const float* pm_K     = (const float*)d_in[4];
  const float* pm_V     = (const float*)d_in[5];
  const float* pm_a     = (const float*)d_in[6];
  const float* eK0      = (const float*)d_in[7];
  const float* eV0      = (const float*)d_in[8];
  const float* W_k      = (const float*)d_in[9];
  const float* b_k      = (const float*)d_in[10];
  const float* W_v      = (const float*)d_in[11];
  const float* b_v      = (const float*)d_in[12];

  float* y_out  = (float*)d_out;                           // [BS,P,D]
  float* eK_out = y_out  + (size_t)BS * P_TOK * D_DIM;     // [BS,R,D]
  float* eV_out = eK_out + (size_t)BS * R_SLOT * D_DIM;    // [BS,R,D]

  float* Kraw = (float*)d_ws;                              // [4096,1024]
  float* Vraw = Kraw + (size_t)M_TOTAL * D_DIM;            // [4096,1024]

  dim3 g1(D_DIM / 128, M_TOTAL / 128, 2);
  gemm_bf16x3_kernel<<<g1, 256, 0, stream>>>(x_all, W_k, b_k, Kraw,
                                             h_all, W_v, b_v, Vraw);

  elig_kernel<<<dim3(BS), 256, 0, stream>>>(Kraw, Vraw, surprise, reset,
                                            eK0, eV0, eK_out, eV_out);

  readout_kernel<<<dim3(BS, P_TOK / 8), 256, 0, stream>>>(x_all, pm_K, pm_V,
                                                          pm_a, y_out);
}
